// ScaphoidAdaPoinTr_83468394430911
// MI455X (gfx1250) — matrix operand order repaired and checked by reading, not compiled, not run on hardware
//
#include <hip/hip_runtime.h>
#include <math.h>

// ---------------------------------------------------------------------------
// Types for WMMA
// ---------------------------------------------------------------------------
typedef _Float16 f16_t;
typedef __attribute__((ext_vector_type(16))) _Float16 v16h;
typedef __attribute__((ext_vector_type(8)))  float    v8f;
typedef __attribute__((ext_vector_type(4)))  float    f32x4;

#define ACT_NONE  0
#define ACT_GELU  1
#define ACT_LRELU 2

// Full (unchecked) A-fragment: lane holds M=lane&15, K chunk {kb..kb+15}
static __device__ __forceinline__ v16h load_a_full(const float* __restrict__ Ap,
                                                   int k0, int vecA)
{
    v16h a;
    if (vecA) {
        f32x4 a0 = *(const f32x4*)(Ap + k0);
        f32x4 a1 = *(const f32x4*)(Ap + k0 + 4);
        f32x4 a2 = *(const f32x4*)(Ap + k0 + 8);
        f32x4 a3 = *(const f32x4*)(Ap + k0 + 12);
#pragma unroll
        for (int i = 0; i < 4; ++i) {
            a[i]      = (f16_t)a0[i];
            a[4 + i]  = (f16_t)a1[i];
            a[8 + i]  = (f16_t)a2[i];
            a[12 + i] = (f16_t)a3[i];
        }
    } else {
#pragma unroll
        for (int i = 0; i < 8; ++i) {
            a[i]     = (f16_t)Ap[k0 + i];
            a[8 + i] = (f16_t)Ap[k0 + 8 + i];
        }
    }
    return a;
}

// K-tail A-fragment: branch-free clamped loads + select (no exec-mask churn).
// Ap is pre-offset by mA*lda + kbA; element e lives at Ap[e - kbA]; clamped
// index stays within the (valid) row since lda >= K.
static __device__ __forceinline__ v16h load_a_tail(const float* __restrict__ Ap,
                                                   int k0, int K, int kbA)
{
    v16h a;
#pragma unroll
    for (int i = 0; i < 8; ++i) {
        int kk  = k0 + kbA + i;
        int kc  = (kk < K) ? kk : (K - 1);
        float v0 = Ap[kc - kbA];
        a[i] = (kk < K) ? (f16_t)v0 : (f16_t)0.0f;
        int kk2 = kk + 8;
        int kc2 = (kk2 < K) ? kk2 : (K - 1);
        float v1 = Ap[kc2 - kbA];
        a[8 + i] = (kk2 < K) ? (f16_t)v1 : (f16_t)0.0f;
    }
    return a;
}

// 16 contiguous K values (transB fast path B fragment)
static __device__ __forceinline__ v16h load_b_contig(const float* __restrict__ Bq,
                                                     int k0, int vecB)
{
    v16h b;
    if (vecB) {
        f32x4 b0 = *(const f32x4*)(Bq + k0);
        f32x4 b1 = *(const f32x4*)(Bq + k0 + 4);
        f32x4 b2 = *(const f32x4*)(Bq + k0 + 8);
        f32x4 b3 = *(const f32x4*)(Bq + k0 + 12);
#pragma unroll
        for (int i = 0; i < 4; ++i) {
            b[i]      = (f16_t)b0[i];
            b[4 + i]  = (f16_t)b1[i];
            b[8 + i]  = (f16_t)b2[i];
            b[12 + i] = (f16_t)b3[i];
        }
    } else {
#pragma unroll
        for (int i = 0; i < 16; ++i) b[i] = (f16_t)Bq[k0 + i];
    }
    return b;
}

// ---------------------------------------------------------------------------
// Batched WMMA GEMM:  C = act(alpha * A @ B + bias)
//   A: (M,K) row-major, row stride lda
//   B: transB==0 -> B[k,n] = Bp[k*ldb+n] ; transB==1 -> B[k,n] = Bp[n*ldb+k]
//   Two-level batch (z = z1*nb2 + z2) with element strides per operand.
// 128 threads = 4 waves; block owns a 32x32 macro tile (2x2 of 16x16 WMMA).
// Fast path (block-uniform full tile): wide unconditional loads, weight
// operand staged through LDS (transB=0) or dual-accumulator K-unroll
// (transB=1).  All checked loads are clamped-address + select (branch-free).
// ---------------------------------------------------------------------------
__global__ __launch_bounds__(128)
void k_gemm(const float* __restrict__ A, const float* __restrict__ Bp,
            const float* __restrict__ bias, float* __restrict__ C,
            int M, int N, int K, int lda, int ldb, int ldc,
            int transB, float alpha, int act, int nb2,
            long long sA1, long long sA2, long long sB1, long long sB2,
            long long sC1, long long sC2, int vecA, int vecB)
{
    __shared__ float ldsB[32 * 33];

    int z = blockIdx.z;
    int z1 = z / nb2, z2 = z % nb2;
    A  += z1 * sA1 + z2 * sA2;
    Bp += z1 * sB1 + z2 * sB2;
    C  += z1 * sC1 + z2 * sC2;

    int bM0 = blockIdx.y * 32, bN0 = blockIdx.x * 32;
    int wave = threadIdx.x >> 5;
    int lane = threadIdx.x & 31;
    int tileM = bM0 + (wave >> 1) * 16;
    int tileN = bN0 + (wave & 1) * 16;

    int mA  = tileM + (lane & 15);
    int kbA = (lane < 16) ? 0 : 16;
    int nB  = tileN + (lane & 15);
    int kbB = (lane < 16) ? 0 : 16;

    v8f acc = {};
    bool fast = (bM0 + 32 <= M) && (bN0 + 32 <= N);   // block-uniform

    if (fast) {
        const float* Ap = A + (long long)mA * lda + kbA;
        if (!transB) {
            // cooperative LDS staging of the 32(K) x 32(N) B tile
            int sk = threadIdx.x >> 2;          // 0..31 : k row
            int sn = (threadIdx.x & 3) * 8;     // 0,8,16,24 : n chunk
            const float* Bs = Bp + bN0 + sn;
            int col = ((wave & 1) << 4) + (lane & 15);   // 0..31 in block
            for (int k0 = 0; k0 < K; k0 += 32) {
                bool full = (k0 + 32 <= K);
                int krow = k0 + sk;
                int krcl = (krow < K) ? krow : (K - 1);    // clamped, always valid
                const float* src = Bs + (long long)krcl * ldb;
                float tmp[8];
                if (vecB) {
                    f32x4 t0 = *(const f32x4*)(src);
                    f32x4 t1 = *(const f32x4*)(src + 4);
#pragma unroll
                    for (int i = 0; i < 4; ++i) { tmp[i] = t0[i]; tmp[4 + i] = t1[i]; }
                } else {
#pragma unroll
                    for (int i = 0; i < 8; ++i) tmp[i] = src[i];
                }
#pragma unroll
                for (int i = 0; i < 8; ++i)
                    ldsB[sk * 33 + sn + i] = (krow < K) ? tmp[i] : 0.0f;

                // A fragment load issued before the barrier to overlap latency
                v16h a = full ? load_a_full(Ap, k0, vecA)
                              : load_a_tail(Ap, k0, K, kbA);
                __syncthreads();

                v16h b;
#pragma unroll
                for (int i = 0; i < 16; ++i) b[i] = (f16_t)ldsB[(kbB + i) * 33 + col];

                acc = __builtin_amdgcn_wmma_f32_16x16x32_f16(false, a, false, b,
                                                             (short)0, acc, false, false);
                __syncthreads();
            }
        } else {
            // transB: per-lane K-contiguous loads, unroll x2 with independent
            // accumulators (hides WMMA->WMMA RAW hazard)
            const float* Bq = Bp + (long long)nB * ldb + kbB;
            v8f acc1 = {};
            int k0 = 0;
            for (; k0 + 64 <= K; k0 += 64) {
                v16h a0 = load_a_full(Ap, k0, vecA);
                v16h b0 = load_b_contig(Bq, k0, vecB);
                v16h a1 = load_a_full(Ap, k0 + 32, vecA);
                v16h b1 = load_b_contig(Bq, k0 + 32, vecB);
                acc  = __builtin_amdgcn_wmma_f32_16x16x32_f16(false, a0, false, b0,
                                                              (short)0, acc, false, false);
                acc1 = __builtin_amdgcn_wmma_f32_16x16x32_f16(false, a1, false, b1,
                                                              (short)0, acc1, false, false);
            }
            for (; k0 < K; k0 += 32) {
                bool full = (k0 + 32 <= K);
                v16h a = full ? load_a_full(Ap, k0, vecA)
                              : load_a_tail(Ap, k0, K, kbA);
                v16h b;
                if (full) {
                    b = load_b_contig(Bq, k0, vecB);
                } else {
#pragma unroll
                    for (int i = 0; i < 16; ++i) {
                        int kk = k0 + kbB + i;
                        int kc = (kk < K) ? kk : (K - 1);
                        float v = Bq[kc - kbB];
                        b[i] = (kk < K) ? (f16_t)v : (f16_t)0.0f;
                    }
                }
                acc = __builtin_amdgcn_wmma_f32_16x16x32_f16(false, a, false, b,
                                                             (short)0, acc, false, false);
            }
#pragma unroll
            for (int i = 0; i < 8; ++i) acc[i] += acc1[i];
        }
    } else {
        // slow path (edge tiles: tiny N / M==16 GEMMs); clamped branch-free loads
        bool mAok = (mA < M);
        bool nBok = (nB < N);
        const float* Arow = A + (long long)(mAok ? mA : 0) * lda;
        long long nOff = (long long)(nBok ? nB : 0);
        for (int k0 = 0; k0 < K; k0 += 32) {
            v16h a;
#pragma unroll
            for (int i = 0; i < 8; ++i) {
                int kk = k0 + kbA + i;
                int kc = (kk < K) ? kk : (K - 1);
                float v0 = Arow[kc];
                a[i] = (mAok && kk < K) ? (f16_t)v0 : (f16_t)0.0f;
                int kk2 = kk + 8;
                int kc2 = (kk2 < K) ? kk2 : (K - 1);
                float v1 = Arow[kc2];
                a[8 + i] = (mAok && kk2 < K) ? (f16_t)v1 : (f16_t)0.0f;
            }
            v16h b;
#pragma unroll
            for (int i = 0; i < 16; ++i) {
                int kk = k0 + kbB + i;
                int kc = (kk < K) ? kk : (K - 1);
                float v = transB ? Bp[nOff * ldb + kc] : Bp[(long long)kc * ldb + nOff];
                b[i] = (nBok && kk < K) ? (f16_t)v : (f16_t)0.0f;
            }
            acc = __builtin_amdgcn_wmma_f32_16x16x32_f16(false, a, false, b,
                                                         (short)0, acc, false, false);
        }
    }

    int n = tileN + (lane & 15);
    if (n < N) {
        float bv = bias ? bias[n] : 0.0f;
#pragma unroll
        for (int v = 0; v < 8; ++v) {
            int m = tileM + v + ((lane >> 4) << 3);
            if (m < M) {
                float x = acc[v] * alpha + bv;
                if (act == ACT_GELU)       x = 0.5f * x * (1.0f + erff(x * 0.70710678118f));
                else if (act == ACT_LRELU) x = (x > 0.0f) ? x : 0.2f * x;
                C[(long long)m * ldc + n] = x;
            }
        }
    }
}

// ---------------------------------------------------------------------------
// kNN (k=16), one thread per query point; register insertion sort
// ---------------------------------------------------------------------------
__global__ void k_knn(const float* __restrict__ ref, const float* __restrict__ qry,
                      int* __restrict__ idx, int Bn, int Nref, int Mq)
{
    int t = blockIdx.x * blockDim.x + threadIdx.x;
    if (t >= Bn * Mq) return;
    int b = t / Mq;
    const float* rp = ref + (long long)b * Nref * 3;
    float qx = qry[(long long)t * 3 + 0];
    float qy = qry[(long long)t * 3 + 1];
    float qz = qry[(long long)t * 3 + 2];

    float bd[16]; int bi[16];
#pragma unroll
    for (int i = 0; i < 16; ++i) { bd[i] = 3.4e38f; bi[i] = 0; }

    for (int nn = 0; nn < Nref; ++nn) {
        float dx = rp[nn * 3 + 0] - qx;
        float dy = rp[nn * 3 + 1] - qy;
        float dz = rp[nn * 3 + 2] - qz;
        float d = dx * dx + dy * dy + dz * dz;
        if (d < bd[15]) {
#pragma unroll
            for (int j = 15; j > 0; --j) {
                bool mv = bd[j - 1] > d;
                if (mv) { bd[j] = bd[j - 1]; bi[j] = bi[j - 1]; }
            }
#pragma unroll
            for (int j = 15; j >= 0; --j) {
                bool place = ((j == 0) || (bd[j - 1] <= d)) && (bd[j] > d);
                if (place) { bd[j] = d; bi[j] = nn; }
            }
        }
    }
#pragma unroll
    for (int i = 0; i < 16; ++i) idx[(long long)t * 16 + i] = bi[i];
}

// ---------------------------------------------------------------------------
// Farthest point sampling: one block per batch
// ---------------------------------------------------------------------------
__global__ void k_fps(const float* __restrict__ xyz, int N, int nSamp,
                      int* __restrict__ out, float* __restrict__ distWs)
{
    int b = blockIdx.x;
    const float* xp = xyz + (long long)b * N * 3;
    float* dist = distWs + (long long)b * N;
    __shared__ float sv[256];
    __shared__ int   si[256];
    __shared__ float cp[3];
    int tid = threadIdx.x;

    if (tid == 0) {
        out[(long long)b * nSamp] = 0;
        cp[0] = xp[0]; cp[1] = xp[1]; cp[2] = xp[2];
    }
    __syncthreads();
    for (int i = tid; i < N; i += 256) {
        float dx = xp[i * 3] - cp[0], dy = xp[i * 3 + 1] - cp[1], dz = xp[i * 3 + 2] - cp[2];
        dist[i] = dx * dx + dy * dy + dz * dz;
    }
    __syncthreads();

    for (int it = 1; it < nSamp; ++it) {
        float best = -1.0f; int bidx = 0x7fffffff;
        for (int i = tid; i < N; i += 256) {
            float d = dist[i];
            if (d > best) { best = d; bidx = i; }
        }
        sv[tid] = best; si[tid] = bidx;
        __syncthreads();
        for (int s = 128; s > 0; s >>= 1) {
            if (tid < s) {
                if (sv[tid + s] > sv[tid] ||
                    (sv[tid + s] == sv[tid] && si[tid + s] < si[tid])) {
                    sv[tid] = sv[tid + s]; si[tid] = si[tid + s];
                }
            }
            __syncthreads();
        }
        if (tid == 0) {
            int nx = si[0];
            out[(long long)b * nSamp + it] = nx;
            cp[0] = xp[nx * 3]; cp[1] = xp[nx * 3 + 1]; cp[2] = xp[nx * 3 + 2];
        }
        __syncthreads();
        for (int i = tid; i < N; i += 256) {
            float dx = xp[i * 3] - cp[0], dy = xp[i * 3 + 1] - cp[1], dz = xp[i * 3 + 2] - cp[2];
            float d = dx * dx + dy * dy + dz * dz;
            if (d < dist[i]) dist[i] = d;
        }
        __syncthreads();
    }
}

// ---------------------------------------------------------------------------
// Gather rows with index: out[b,i,:] = src[b, idx[b,i], :]
// ---------------------------------------------------------------------------
__global__ void k_gather(const float* __restrict__ src, const int* __restrict__ idx,
                         float* __restrict__ out, int Bn, int Nout, int C,
                         long long srcStrideB, long long outStrideB)
{
    long long t = (long long)blockIdx.x * 256 + threadIdx.x;
    long long total = (long long)Bn * Nout * C;
    if (t >= total) return;
    int c = (int)(t % C);
    long long r = t / C;
    int i = (int)(r % Nout);
    int b = (int)(r / Nout);
    int id = idx[(long long)b * Nout + i];
    out[b * outStrideB + (long long)i * C + c] = src[b * srcStrideB + (long long)id * C + c];
}

// feat[b,m,k, 0:C]   = x_k[b, idx, :] - x_q[b,m,:]
// feat[b,m,k, C:2C]  = x_q[b,m,:]
__global__ void k_graph_feat(const float* __restrict__ xk, const float* __restrict__ xq,
                             const int* __restrict__ idx, float* __restrict__ feat,
                             int Bn, int M, int Nk, int C)
{
    long long t = (long long)blockIdx.x * 256 + threadIdx.x;
    long long total = (long long)Bn * M * 16 * 2 * C;
    if (t >= total) return;
    int c = (int)(t % (2 * C));
    long long r = t / (2 * C);
    int k = (int)(r % 16);
    long long bm = r / 16;
    int b = (int)(bm / M);
    if (c < C) {
        int nb = idx[bm * 16 + k];
        float kv = xk[((long long)b * Nk + nb) * C + c];
        feat[t] = kv - xq[bm * C + c];
    } else {
        feat[t] = xq[bm * C + (c - C)];
    }
}

// GroupNorm stats per (b, group): blockIdx.x = b*groups+g, over R rows x (C/groups)
__global__ void k_gn_stats(const float* __restrict__ h, float* __restrict__ stats,
                           int R, int C, int groups)
{
    int b = blockIdx.x / groups, g = blockIdx.x % groups;
    int cg = C / groups;
    long long cnt = (long long)R * cg;
    __shared__ float s1[256], s2[256];
    float a1 = 0.0f, a2 = 0.0f;
    for (long long t = threadIdx.x; t < cnt; t += 256) {
        long long r = t / cg;
        int c = g * cg + (int)(t % cg);
        float v = h[((long long)b * R + r) * C + c];
        a1 += v; a2 += v * v;
    }
    s1[threadIdx.x] = a1; s2[threadIdx.x] = a2;
    __syncthreads();
    for (int s = 128; s > 0; s >>= 1) {
        if (threadIdx.x < s) { s1[threadIdx.x] += s1[threadIdx.x + s]; s2[threadIdx.x] += s2[threadIdx.x + s]; }
        __syncthreads();
    }
    if (threadIdx.x == 0) {
        float mean = s1[0] / (float)cnt;
        float var = s2[0] / (float)cnt - mean * mean;
        stats[blockIdx.x * 2] = mean;
        stats[blockIdx.x * 2 + 1] = rsqrtf(var + 1e-5f);
    }
}

// normalize + affine + LeakyReLU(0.2) + max over k=16 neighbors
__global__ void k_gn_max(const float* __restrict__ h, const float* __restrict__ stats,
                         const float* __restrict__ ga, const float* __restrict__ be,
                         float* __restrict__ out, int Bn, int M, int C, int groups)
{
    long long t = (long long)blockIdx.x * 256 + threadIdx.x;
    long long total = (long long)Bn * M * C;
    if (t >= total) return;
    int c = (int)(t % C);
    long long bm = t / C;
    int b = (int)(bm / M);
    int g = c / (C / groups);
    float mean = stats[(b * groups + g) * 2];
    float rstd = stats[(b * groups + g) * 2 + 1];
    float gam = ga[c], bet = be[c];
    float best = -3.4e38f;
    for (int k = 0; k < 16; ++k) {
        float v = h[(bm * 16 + k) * C + c];
        v = (v - mean) * rstd * gam + bet;
        v = (v > 0.0f) ? v : 0.2f * v;
        best = fmaxf(best, v);
    }
    out[t] = best;
}

// LayerNorm: one block per row
__global__ void k_layernorm(const float* __restrict__ x, const float* __restrict__ g,
                            const float* __restrict__ b, float* __restrict__ y, int C)
{
    long long row = blockIdx.x;
    const float* xr = x + row * C;
    float* yr = y + row * C;
    __shared__ float red[128];
    int tid = threadIdx.x;
    float s = 0.0f;
    for (int i = tid; i < C; i += 128) s += xr[i];
    red[tid] = s; __syncthreads();
    for (int k = 64; k > 0; k >>= 1) { if (tid < k) red[tid] += red[tid + k]; __syncthreads(); }
    float mean = red[0] / (float)C;
    __syncthreads();
    s = 0.0f;
    for (int i = tid; i < C; i += 128) { float d = xr[i] - mean; s += d * d; }
    red[tid] = s; __syncthreads();
    for (int k = 64; k > 0; k >>= 1) { if (tid < k) red[tid] += red[tid + k]; __syncthreads(); }
    float rstd = rsqrtf(red[0] / (float)C + 1e-5f);
    for (int i = tid; i < C; i += 128) yr[i] = (xr[i] - mean) * rstd * g[i] + b[i];
}

// row softmax (block per row)
__global__ void k_softmax(float* __restrict__ x, int L)
{
    float* row = x + (long long)blockIdx.x * L;
    __shared__ float red[128];
    int tid = threadIdx.x;
    float mx = -3.4e38f;
    for (int i = tid; i < L; i += 128) mx = fmaxf(mx, row[i]);
    red[tid] = mx; __syncthreads();
    for (int s = 64; s > 0; s >>= 1) { if (tid < s) red[tid] = fmaxf(red[tid], red[tid + s]); __syncthreads(); }
    mx = red[0]; __syncthreads();
    float sum = 0.0f;
    for (int i = tid; i < L; i += 128) { float e = expf(row[i] - mx); row[i] = e; sum += e; }
    red[tid] = sum; __syncthreads();
    for (int s = 64; s > 0; s >>= 1) { if (tid < s) red[tid] += red[tid + s]; __syncthreads(); }
    float inv = 1.0f / red[0];
    for (int i = tid; i < L; i += 128) row[i] *= inv;
}

// max over axis 1 of (B,T,C) -> (B,C)
__global__ void k_rowmax(const float* __restrict__ x, float* __restrict__ out,
                         int Bn, int T, int C)
{
    int t = blockIdx.x * 256 + threadIdx.x;
    if (t >= Bn * C) return;
    int b = t / C, c = t % C;
    float best = -3.4e38f;
    for (int i = 0; i < T; ++i)
        best = fmaxf(best, x[((long long)b * T + i) * C + c]);
    out[t] = best;
}

__global__ void k_add(float* __restrict__ dst, const float* __restrict__ src, long long n)
{
    long long t = (long long)blockIdx.x * 256 + threadIdx.x;
    if (t < n) dst[t] += src[t];
}

__global__ void k_copyf(float* __restrict__ dst, const float* __restrict__ src, long long n)
{
    long long t = (long long)blockIdx.x * 256 + threadIdx.x;
    if (t < n) dst[t] = src[t];
}

__global__ void k_copy2d(float* __restrict__ dst, long long dStrideB,
                         const float* __restrict__ src, long long sStrideB,
                         int Bn, int per)
{
    long long t = (long long)blockIdx.x * 256 + threadIdx.x;
    if (t >= (long long)Bn * per) return;
    int b = (int)(t / per);
    int i = (int)(t % per);
    dst[b * dStrideB + i] = src[b * sStrideB + i];
}

// BN (eval) + LeakyReLU(0.2), per channel
__global__ void k_bn_leaky(float* __restrict__ x, const float* __restrict__ m,
                           const float* __restrict__ v, const float* __restrict__ g,
                           const float* __restrict__ b, long long rows, int C)
{
    long long t = (long long)blockIdx.x * 256 + threadIdx.x;
    if (t >= rows * C) return;
    int c = (int)(t % C);
    float y = (x[t] - m[c]) * rsqrtf(v[c] + 1e-5f) * g[c] + b[c];
    x[t] = (y > 0.0f) ? y : 0.2f * y;
}

// per-batch descending sort of n<=1024 values (bitonic, pad -inf); top-512 indices
__global__ void k_rank_sort(const float* __restrict__ rank, int* __restrict__ order, int n)
{
    __shared__ float v[1024];
    __shared__ int   id[1024];
    int b = blockIdx.x;
    for (int i = threadIdx.x; i < 1024; i += 256) {
        v[i] = (i < n) ? rank[(long long)b * n + i] : -3.4e38f;
        id[i] = i;
    }
    __syncthreads();
    for (int ksz = 2; ksz <= 1024; ksz <<= 1) {
        for (int j = ksz >> 1; j > 0; j >>= 1) {
            for (int i = threadIdx.x; i < 1024; i += 256) {
                int ixj = i ^ j;
                if (ixj > i) {
                    bool descSeg = ((i & ksz) == 0);
                    bool sw = descSeg ? (v[i] < v[ixj]) : (v[i] > v[ixj]);
                    if (sw) {
                        float tv = v[i]; v[i] = v[ixj]; v[ixj] = tv;
                        int ti = id[i]; id[i] = id[ixj]; id[ixj] = ti;
                    }
                }
            }
            __syncthreads();
        }
    }
    for (int i = threadIdx.x; i < 512; i += 256) order[(long long)b * 512 + i] = id[i];
}

// out row = [ s1[b,:] (d1, per-batch) | s2[row,:] (d2) | s3[row,:] (d3) ]
__global__ void k_concat3(float* __restrict__ out, const float* __restrict__ s1, int d1,
                          const float* __restrict__ s2, int d2,
                          const float* __restrict__ s3, int d3,
                          int tokens, long long rows)
{
    int D = d1 + d2 + d3;
    long long t = (long long)blockIdx.x * 256 + threadIdx.x;
    if (t >= rows * D) return;
    int c = (int)(t % D);
    long long row = t / D;
    long long b = row / tokens;
    float val;
    if (c < d1)            val = s1[b * d1 + c];
    else if (c < d1 + d2)  val = s2[row * d2 + (c - d1)];
    else                   val = s3[row * d3 + (c - d1 - d2)];
    out[t] = val;
}

// rebuild = rel + coarse broadcast over step
__global__ void k_rebuild(float* __restrict__ out, const float* __restrict__ rel,
                          const float* __restrict__ coarse, long long n)
{
    long long t = (long long)blockIdx.x * 256 + threadIdx.x;
    if (t >= n) return;
    int c = (int)(t % 3);
    long long r = t / 3;
    long long q = r / 16;   // b*512 + qi
    out[t] = rel[t] + coarse[q * 3 + c];
}

// ---------------------------------------------------------------------------
// Host helpers
// ---------------------------------------------------------------------------
static inline unsigned blks(long long n) { return (unsigned)((n + 255) / 256); }

static void gemm(hipStream_t st, const float* A, const float* Bm, const float* bias,
                 float* C, int M, int Nn, int K, int lda, int ldb, int ldc,
                 int transB, float alpha, int act,
                 int nb1 = 1, int nb2 = 1,
                 long long sA1 = 0, long long sA2 = 0, long long sB1 = 0,
                 long long sB2 = 0, long long sC1 = 0, long long sC2 = 0)
{
    int vecA = ((lda & 3) == 0) && ((sA1 & 3) == 0) && ((sA2 & 3) == 0) &&
               ((((uintptr_t)A) & 15) == 0);
    int vecB = ((ldb & 3) == 0) && ((sB1 & 3) == 0) && ((sB2 & 3) == 0) &&
               ((((uintptr_t)Bm) & 15) == 0);
    dim3 grid((Nn + 31) / 32, (M + 31) / 32, nb1 * nb2);
    k_gemm<<<grid, dim3(128), 0, st>>>(A, Bm, bias, C, M, Nn, K, lda, ldb, ldc,
                                       transB, alpha, act, nb2,
                                       sA1, sA2, sB1, sB2, sC1, sC2, vecA, vecB);
}

static void self_attn(hipStream_t st, const float* qkv, float* scores, float* out,
                      int Bn, int n, int C, int H)
{
    int d = C / H;
    float alpha = 1.0f / sqrtf((float)d);
    gemm(st, qkv, qkv + C, nullptr, scores, n, n, d, 3 * C, 3 * C, n, 1, alpha, ACT_NONE,
         Bn, H, (long long)n * 3 * C, d, (long long)n * 3 * C, d,
         (long long)H * n * n, (long long)n * n);
    k_softmax<<<dim3(Bn * H * n), dim3(128), 0, st>>>(scores, n);
    gemm(st, scores, qkv + 2 * C, nullptr, out, n, d, n, n, 3 * C, C, 0, 1.0f, ACT_NONE,
         Bn, H, (long long)H * n * n, (long long)n * n, (long long)n * 3 * C, d,
         (long long)n * C, d);
}

static void cross_attn(hipStream_t st, const float* cq, const float* ck, const float* cv,
                       float* scores, float* out, int Bn, int nq, int nv, int C, int H)
{
    int d = C / H;
    float alpha = 1.0f / sqrtf((float)d);
    gemm(st, cq, ck, nullptr, scores, nq, nv, d, C, C, nv, 1, alpha, ACT_NONE,
         Bn, H, (long long)nq * C, d, (long long)nv * C, d,
         (long long)H * nq * nv, (long long)nq * nv);
    k_softmax<<<dim3(Bn * H * nq), dim3(128), 0, st>>>(scores, nv);
    gemm(st, scores, cv, nullptr, out, nq, d, nv, nv, C, C, 0, 1.0f, ACT_NONE,
         Bn, H, (long long)H * nq * nv, (long long)nq * nv, (long long)nv * C, d,
         (long long)nq * C, d);
}

struct Lin { const float* b; const float* w; };
struct LNp { const float* b; const float* g; };
struct Gcv { const float* b; const float* g; const float* w; };
struct EncB { Lin fc1, fc2; LNp ln1, ln2; Lin proj, qkv; };
struct DecB { Lin ck, cproj, cq, cv, fc1, fc2; LNp ln1, ln2, lnq, lnv; Lin proj, qkv; };

extern "C" void kernel_launch(void* const* d_in, const int* in_sizes, int n_in,
                              void* d_out_v, int out_size, void* d_ws, size_t ws_size,
                              hipStream_t st)
{
    (void)out_size; (void)ws_size;
    const int B = 16, N = 2048, H = 6, DIM = 384, MLPH = 768, NQ = 512, GD = 1024;

    // ---- locate xyz & parse params (jax pytree order: sorted dict keys) ----
    int ip = 0;
    const float* xyz;
    if (in_sizes[0] == B * N * 3) { xyz = (const float*)d_in[0]; ip = 1; }
    else                          { xyz = (const float*)d_in[n_in - 1]; ip = 0; }

    auto PL = [&]() { Lin l; l.b = (const float*)d_in[ip++]; l.w = (const float*)d_in[ip++]; return l; };
    auto PN = [&]() { LNp l; l.b = (const float*)d_in[ip++]; l.g = (const float*)d_in[ip++]; return l; };
    auto PG = [&]() { Gcv g; g.b = (const float*)d_in[ip++]; g.g = (const float*)d_in[ip++];
                      g.w = (const float*)d_in[ip++]; return g; };

    Lin cp0 = PL(), cp1 = PL();                  // coarse_pred
    Lin dh0 = PL(), dh1 = PL();                  // decode_head
    DecB dec[8];
    for (int i = 0; i < 8; ++i) {
        dec[i].ck = PL(); dec[i].cproj = PL(); dec[i].cq = PL(); dec[i].cv = PL();
        dec[i].fc1 = PL(); dec[i].fc2 = PL();
        dec[i].ln1 = PN(); dec[i].ln2 = PN(); dec[i].lnq = PN(); dec[i].lnv = PN();
        dec[i].proj = PL(); dec[i].qkv = PL();
    }
    EncB enc[6];
    for (int i = 0; i < 6; ++i) {
        enc[i].fc1 = PL(); enc[i].fc2 = PL();
        enc[i].ln1 = PN(); enc[i].ln2 = PN();
        enc[i].proj = PL(); enc[i].qkv = PL();
    }
    Lin g_it = PL();
    Gcv gl1 = PG(), gl2 = PG(), gl3 = PG(), gl4 = PG();
    const float* bn_b = (const float*)d_in[ip++];
    const float* bn_g = (const float*)d_in[ip++];
    const float* bn_m = (const float*)d_in[ip++];
    const float* bn_v = (const float*)d_in[ip++];
    Lin hc1 = PL(), hc2 = PL();
    Lin inc0 = PL(), inc1 = PL();
    Lin ip0 = PL(), ip1 = PL();
    Lin mq0 = PL(), mq1 = PL(), mq2 = PL();
    Lin pe0 = PL(), pe1 = PL();
    Lin qr0 = PL(), qr1 = PL(), qr2 = PL();
    Lin rmap = PL();

    // ---- workspace allocation (256B-aligned bump allocator) ----
    char* base = (char*)d_ws;
    size_t off = 0;
    auto AF = [&](size_t n) { float* p = (float*)(base + off); off += ((n * 4 + 255) / 256) * 256; return p; };
    auto AI = [&](size_t n) { int* p = (int*)(base + off); off += ((n * 4 + 255) / 256) * 256; return p; };

    int*   idxK    = AI((size_t)B * 2048 * 16);
    int*   i1      = AI(B * 512);
    int*   i2      = AI(B * 128);
    int*   iC      = AI(B * 256);
    int*   order   = AI(B * 512);
    float* fpsd    = AF((size_t)B * 2048);
    float* stats   = AF(256);
    float* featA   = AF((size_t)16777216);   // graph feats; later reused as attn scores
    float* hA      = AF((size_t)16777216);   // graph conv out; scores spill into here too
    float* f0      = AF((size_t)B * 2048 * 8);
    float* l1out   = AF((size_t)B * 2048 * 32);
    float* coor1   = AF((size_t)B * 512 * 3);
    float* f1g     = AF((size_t)B * 512 * 32);
    float* l2out   = AF((size_t)B * 512 * 64);
    float* l3out   = AF((size_t)B * 512 * 64);
    float* coor2   = AF((size_t)B * 128 * 3);
    float* f2g     = AF((size_t)B * 128 * 64);
    float* l4out   = AF((size_t)B * 128 * 128);
    float* xtok    = AF((size_t)B * 128 * DIM);
    float* tA      = AF((size_t)B * 512 * 1152);
    float* tB      = AF((size_t)B * 512 * 1024);
    float* tC      = AF((size_t)B * 512 * 1024);
    float* tD      = AF((size_t)B * 512 * DIM);
    float* gf      = AF((size_t)B * GD);
    float* cA      = AF((size_t)B * 1536);
    float* coarseAll = AF((size_t)B * 768 * 3);
    float* rankbuf = AF((size_t)B * 768);
    float* coarse_sel = AF((size_t)B * 512 * 3);
    float* qtok    = AF((size_t)B * 512 * DIM);
    float* xvln    = AF((size_t)B * 128 * DIM);
    float* cqb     = AF((size_t)B * 512 * DIM);
    float* ckb     = AF((size_t)B * 128 * DIM);
    float* cvb     = AF((size_t)B * 128 * DIM);
    float* g2      = AF((size_t)B * 1024);
    float* rf      = AF((size_t)B * 512 * DIM);
    float* rfmax   = AF((size_t)B * DIM);
    float* rel     = AF((size_t)B * 512 * 48);
    float* mq_in   = AF((size_t)B * 512 * 1027);
    float* rfcat   = AF((size_t)B * 512 * 1411);
    float* pf      = AF((size_t)B * 512 * MLPH);
    // scores (96*512*512 = 25.17M floats) aliases the contiguous featA+hA arena
    // (33.5M floats), both dead after the grouper stage.
    float* scores  = featA;

    float* d_out = (float*)d_out_v;

    // =============================== grouper ===============================
    gemm(st, xyz, g_it.w, g_it.b, f0, B * N, 8, 3, 3, 8, 8, 0, 1.0f, ACT_NONE);

    auto graph_layer = [&](const float* coor_q, const float* xq, int M,
                           const float* coor_k, const float* xk, int Nk,
                           int C, int Cout, const Gcv& p, float* out) {
        int tot = B * M;
        k_knn<<<dim3((tot + 127) / 128), dim3(128), 0, st>>>(coor_k, coor_q, idxK, B, Nk, M);
        long long tf = (long long)B * M * 16 * 2 * C;
        k_graph_feat<<<blks(tf), dim3(256), 0, st>>>(xk, xq, idxK, featA, B, M, Nk, C);
        gemm(st, featA, p.w, nullptr, hA, B * M * 16, Cout, 2 * C, 2 * C, Cout, Cout,
             0, 1.0f, ACT_NONE);
        k_gn_stats<<<dim3(B * 4), dim3(256), 0, st>>>(hA, stats, M * 16, Cout, 4);
        long long tg = (long long)B * M * Cout;
        k_gn_max<<<blks(tg), dim3(256), 0, st>>>(hA, stats, p.g, p.b, out, B, M, Cout, 4);
    };

    graph_layer(xyz, f0, N, xyz, f0, N, 8, 32, gl1, l1out);
    k_fps<<<dim3(B), dim3(256), 0, st>>>(xyz, N, 512, i1, fpsd);
    k_gather<<<blks((long long)B * 512 * 3), dim3(256), 0, st>>>(xyz, i1, coor1, B, 512, 3,
        (long long)N * 3, (long long)512 * 3);
    k_gather<<<blks((long long)B * 512 * 32), dim3(256), 0, st>>>(l1out, i1, f1g, B, 512, 32,
        (long long)N * 32, (long long)512 * 32);
    graph_layer(coor1, f1g, 512, xyz, l1out, N, 32, 64, gl2, l2out);
    graph_layer(coor1, l2out, 512, coor1, l2out, 512, 64, 64, gl3, l3out);
    k_fps<<<dim3(B), dim3(256), 0, st>>>(coor1, 512, 128, i2, fpsd);
    k_gather<<<blks((long long)B * 128 * 3), dim3(256), 0, st>>>(coor1, i2, coor2, B, 128, 3,
        (long long)512 * 3, (long long)128 * 3);
    k_gather<<<blks((long long)B * 128 * 64), dim3(256), 0, st>>>(l3out, i2, f2g, B, 128, 64,
        (long long)512 * 64, (long long)128 * 64);
    graph_layer(coor2, f2g, 128, coor1, l3out, 512, 64, 128, gl4, l4out);

    // =================== input_proj + pos_embed -> tokens ==================
    int rT = B * 128;
    gemm(st, l4out, ip0.w, ip0.b, tB, rT, 512, 128, 128, 512, 512, 0, 1.0f, ACT_GELU);
    gemm(st, tB, ip1.w, ip1.b, xtok, rT, DIM, 512, 512, DIM, DIM, 0, 1.0f, ACT_NONE);
    gemm(st, coor2, pe0.w, pe0.b, tC, rT, 128, 3, 3, 128, 128, 0, 1.0f, ACT_GELU);
    gemm(st, tC, pe1.w, pe1.b, tD, rT, DIM, 128, 128, DIM, DIM, 0, 1.0f, ACT_NONE);
    k_add<<<blks((long long)rT * DIM), dim3(256), 0, st>>>(xtok, tD, (long long)rT * DIM);

    // ============================== encoder ================================
    for (int l = 0; l < 6; ++l) {
        const EncB& e = enc[l];
        k_layernorm<<<dim3(rT), dim3(128), 0, st>>>(xtok, e.ln1.g, e.ln1.b, tD, DIM);
        gemm(st, tD, e.qkv.w, e.qkv.b, tA, rT, 3 * DIM, DIM, DIM, 3 * DIM, 3 * DIM, 0, 1.0f, ACT_NONE);
        self_attn(st, tA, scores, tD, B, 128, DIM, H);
        gemm(st, tD, e.proj.w, e.proj.b, tC, rT, DIM, DIM, DIM, DIM, DIM, 0, 1.0f, ACT_NONE);
        k_add<<<blks((long long)rT * DIM), dim3(256), 0, st>>>(xtok, tC, (long long)rT * DIM);
        k_layernorm<<<dim3(rT), dim3(128), 0, st>>>(xtok, e.ln2.g, e.ln2.b, tD, DIM);
        gemm(st, tD, e.fc1.w, e.fc1.b, tB, rT, MLPH, DIM, DIM, MLPH, MLPH, 0, 1.0f, ACT_GELU);
        gemm(st, tB, e.fc2.w, e.fc2.b, tC, rT, DIM, MLPH, MLPH, DIM, DIM, 0, 1.0f, ACT_NONE);
        k_add<<<blks((long long)rT * DIM), dim3(256), 0, st>>>(xtok, tC, (long long)rT * DIM);
    }

    // ================= global feature / coarse prediction ==================
    gemm(st, xtok, inc0.w, inc0.b, tB, rT, 1024, DIM, DIM, 1024, 1024, 0, 1.0f, ACT_GELU);
    gemm(st, tB, inc1.w, inc1.b, tC, rT, GD, 1024, 1024, GD, GD, 0, 1.0f, ACT_NONE);
    k_rowmax<<<blks(B * GD), dim3(256), 0, st>>>(tC, gf, B, 128, GD);

    gemm(st, gf, cp0.w, cp0.b, tB, B, 1024, GD, GD, 1024, 1024, 0, 1.0f, ACT_GELU);
    gemm(st, tB, cp1.w, cp1.b, cA, B, 1536, 1024, 1024, 1536, 1536, 0, 1.0f, ACT_NONE);

    k_fps<<<dim3(B), dim3(256), 0, st>>>(xyz, N, 256, iC, fpsd);
    k_copy2d<<<blks((long long)B * 1536), dim3(256), 0, st>>>(coarseAll, 2304, cA, 1536, B, 1536);
    k_gather<<<blks((long long)B * 256 * 3), dim3(256), 0, st>>>(xyz, iC, coarseAll + 1536,
        B, 256, 3, (long long)N * 3, (long long)2304);

    // ======================= query ranking + select ========================
    int rR = B * 768;
    gemm(st, coarseAll, qr0.w, qr0.b, tB, rR, 256, 3, 3, 256, 256, 0, 1.0f, ACT_GELU);
    gemm(st, tB, qr1.w, qr1.b, tC, rR, 256, 256, 256, 256, 256, 0, 1.0f, ACT_GELU);
    gemm(st, tC, qr2.w, qr2.b, rankbuf, rR, 1, 256, 256, 1, 1, 0, 1.0f, ACT_NONE);
    // sigmoid is monotone -> ordering unchanged, sort raw logits
    k_rank_sort<<<dim3(B), dim3(256), 0, st>>>(rankbuf, order, 768);
    k_gather<<<blks((long long)B * 512 * 3), dim3(256), 0, st>>>(coarseAll, order, coarse_sel,
        B, 512, 3, (long long)2304, (long long)512 * 3);

    // ============================= mlp_query ===============================
    int rQ = B * NQ;
    k_concat3<<<blks((long long)rQ * 1027), dim3(256), 0, st>>>(mq_in, gf, GD,
        coarse_sel, 3, nullptr, 0, NQ, (long long)rQ);
    gemm(st, mq_in, mq0.w, mq0.b, tB, rQ, 1024, 1027, 1027, 1024, 1024, 0, 1.0f, ACT_GELU);
    gemm(st, tB, mq1.w, mq1.b, tC, rQ, 1024, 1024, 1024, 1024, 1024, 0, 1.0f, ACT_GELU);
    gemm(st, tC, mq2.w, mq2.b, qtok, rQ, DIM, 1024, 1024, DIM, DIM, 0, 1.0f, ACT_NONE);

    // ============================== decoder ================================
    for (int l = 0; l < 8; ++l) {
        const DecB& d = dec[l];
        k_layernorm<<<dim3(rQ), dim3(128), 0, st>>>(qtok, d.ln1.g, d.ln1.b, tD, DIM);
        gemm(st, tD, d.qkv.w, d.qkv.b, tA, rQ, 3 * DIM, DIM, DIM, 3 * DIM, 3 * DIM, 0, 1.0f, ACT_NONE);
        self_attn(st, tA, scores, tD, B, NQ, DIM, H);
        gemm(st, tD, d.proj.w, d.proj.b, tC, rQ, DIM, DIM, DIM, DIM, DIM, 0, 1.0f, ACT_NONE);
        k_add<<<blks((long long)rQ * DIM), dim3(256), 0, st>>>(qtok, tC, (long long)rQ * DIM);

        k_layernorm<<<dim3(rQ), dim3(128), 0, st>>>(qtok, d.lnq.g, d.lnq.b, tD, DIM);
        k_layernorm<<<dim3(rT), dim3(128), 0, st>>>(xtok, d.lnv.g, d.lnv.b, xvln, DIM);
        gemm(st, tD, d.cq.w, d.cq.b, cqb, rQ, DIM, DIM, DIM, DIM, DIM, 0, 1.0f, ACT_NONE);
        gemm(st, xvln, d.ck.w, d.ck.b, ckb, rT, DIM, DIM, DIM, DIM, DIM, 0, 1.0f, ACT_NONE);
        gemm(st, xvln, d.cv.w, d.cv.b, cvb, rT, DIM, DIM, DIM, DIM, DIM, 0, 1.0f, ACT_NONE);
        cross_attn(st, cqb, ckb, cvb, scores, tD, B, NQ, 128, DIM, H);
        gemm(st, tD, d.cproj.w, d.cproj.b, tC, rQ, DIM, DIM, DIM, DIM, DIM, 0, 1.0f, ACT_NONE);
        k_add<<<blks((long long)rQ * DIM), dim3(256), 0, st>>>(qtok, tC, (long long)rQ * DIM);

        k_layernorm<<<dim3(rQ), dim3(128), 0, st>>>(qtok, d.ln2.g, d.ln2.b, tD, DIM);
        gemm(st, tD, d.fc1.w, d.fc1.b, tB, rQ, MLPH, DIM, DIM, MLPH, MLPH, 0, 1.0f, ACT_GELU);
        gemm(st, tB, d.fc2.w, d.fc2.b, tC, rQ, DIM, MLPH, MLPH, DIM, DIM, 0, 1.0f, ACT_NONE);
        k_add<<<blks((long long)rQ * DIM), dim3(256), 0, st>>>(qtok, tC, (long long)rQ * DIM);
    }

    // ============================ rebuild head =============================
    gemm(st, qtok, hc1.w, hc1.b, tB, rQ, 1024, DIM, DIM, 1024, 1024, 0, 1.0f, ACT_NONE);
    k_bn_leaky<<<blks((long long)rQ * 1024), dim3(256), 0, st>>>(tB, bn_m, bn_v, bn_g, bn_b,
                                                                 (long long)rQ, 1024);
    gemm(st, tB, hc2.w, hc2.b, tC, rQ, 1024, 1024, 1024, 1024, 1024, 0, 1.0f, ACT_NONE);
    k_rowmax<<<blks(B * 1024), dim3(256), 0, st>>>(tC, g2, B, NQ, 1024);

    k_concat3<<<blks((long long)rQ * 1411), dim3(256), 0, st>>>(rfcat, g2, 1024,
        qtok, DIM, coarse_sel, 3, NQ, (long long)rQ);
    gemm(st, rfcat, rmap.w, rmap.b, rf, rQ, DIM, 1411, 1411, DIM, DIM, 0, 1.0f, ACT_NONE);
    k_rowmax<<<blks(B * DIM), dim3(256), 0, st>>>(rf, rfmax, B, NQ, DIM);
    k_concat3<<<blks((long long)rQ * 768), dim3(256), 0, st>>>(pf, rfmax, DIM,
        rf, DIM, nullptr, 0, NQ, (long long)rQ);
    gemm(st, pf, dh0.w, dh0.b, tB, rQ, 512, 768, 768, 512, 512, 0, 1.0f, ACT_GELU);
    gemm(st, tB, dh1.w, dh1.b, rel, rQ, 48, 512, 512, 48, 48, 0, 1.0f, ACT_NONE);

    long long nCoarse = (long long)B * 512 * 3;
    long long nReb = (long long)B * 8192 * 3;
    k_copyf<<<blks(nCoarse), dim3(256), 0, st>>>(d_out, coarse_sel, nCoarse);
    k_rebuild<<<blks(nReb), dim3(256), 0, st>>>(d_out + nCoarse, rel, coarse_sel, nReb);
}
